// HMM_9912784519224
// MI455X (gfx1250) — matrix operand order repaired and checked by reading, not compiled, not run on hardware
//
#include <hip/hip_runtime.h>
#include <hip/hip_bf16.h>
#include <math.h>

// ---- problem constants (match reference setup_inputs) ----
#define KS  512          // number of HMM states
#define TT  1024         // time steps
#define BB  64           // batch
#define BG  16           // batches per workgroup (= WMMA M tile)
#define NEG_INF      (-3.402823466e38f)
#define HALF_LOG_2PI (0.9189385332046727f)

typedef __attribute__((ext_vector_type(16))) _Float16 v16h;
typedef __attribute__((ext_vector_type(8)))  _Float16 v8h;
typedef __attribute__((ext_vector_type(8)))  float    v8f;

__device__ __forceinline__ float wave_max32(float v) {
#pragma unroll
    for (int m = 16; m >= 1; m >>= 1) v = fmaxf(v, __shfl_xor(v, m, 32));
    return v;
}
__device__ __forceinline__ float wave_sum32(float v) {
#pragma unroll
    for (int m = 16; m >= 1; m >>= 1) v += __shfl_xor(v, m, 32);
    return v;
}

// ---------------------------------------------------------------------------
// Precompute EA[i][j] = exp(A[i][j]) (f16, row major) and its transpose EAT.
// Forward step uses EAT (contraction over i contiguous), backward uses EA.
// exp(N(0,1)) fits comfortably in f16 range; f16's 10 mantissa bits beat bf16.
// ---------------------------------------------------------------------------
__global__ void hmm_prep_expA(const float* __restrict__ A,
                              _Float16* __restrict__ EAT,
                              _Float16* __restrict__ EA) {
    int idx = blockIdx.x * blockDim.x + threadIdx.x;
    if (idx >= KS * KS) return;
    int i = idx / KS, j = idx % KS;
    _Float16 v = (_Float16)__expf(A[idx]);
    EA[idx]         = v;  // EA[i][j]
    EAT[j * KS + i] = v;  // EAT[j][i]
}

// ---------------------------------------------------------------------------
// Forward pass: alpha_t = log(exp(alpha_{t-1}-m) @ exp(A)) + m + e_t
// One workgroup of 512 threads (16 waves) per 16-batch group.
// Wave w: owns batch-row w for max/exp phases; owns column tiles 2w,2w+1 for WMMA.
// alpha is written to d_out (later overwritten by gamma in the backward pass).
// ---------------------------------------------------------------------------
__global__ __launch_bounds__(512) void hmm_forward(
    const float* __restrict__ obvs,
    const float* __restrict__ log_init,
    const float* __restrict__ means,
    const float* __restrict__ log_stds,
    const _Float16* __restrict__ BT,    // EAT: [j][i] = exp(A[i][j])
    float* __restrict__ alpha_out)      // [B][T][K]
{
    __shared__ float    Af[BG][KS];     // current alpha (fp32)
    __shared__ _Float16 Pl[BG][KS];     // exp(alpha - rowmax) (f16)
    __shared__ float    rowm[BG];
    __shared__ float    s_mu[KS], s_inv[KS], s_ct[KS];

    const int tid  = threadIdx.x;
    const int wave = tid >> 5;
    const int lane = tid & 31;
    const int b0   = blockIdx.x * BG;
    const int half = lane >> 4;         // 0: lanes 0-15, 1: lanes 16-31
    const int nidx = lane & 15;

    // cache Gaussian emission params
    {
        float mu = means[tid], ls = log_stds[tid];
        s_mu[tid]  = mu;
        s_inv[tid] = __expf(-2.0f * ls);
        s_ct[tid]  = -ls - HALF_LOG_2PI;
    }
    __syncthreads();

    // t = 0 : alpha0 = emission + log_init
    {
        float li = log_init[tid];
#pragma unroll 4
        for (int b = 0; b < BG; ++b) {
            float x  = obvs[(size_t)(b0 + b) * TT];
            float d  = x - s_mu[tid];
            float a0 = -0.5f * d * d * s_inv[tid] + s_ct[tid] + li;
            Af[b][tid] = a0;
            alpha_out[(size_t)(b0 + b) * TT * KS + tid] = a0;
        }
    }

    for (int t = 1; t < TT; ++t) {
        __syncthreads();
        // ---- phase 1: per-row max + P = exp(alpha - max) (wave w -> row w)
        {
            float mx = NEG_INF;
            for (int k = lane; k < KS; k += 32) mx = fmaxf(mx, Af[wave][k]);
            mx = wave_max32(mx);
            if (lane == 0) rowm[wave] = mx;
            for (int k = lane; k < KS; k += 32)
                Pl[wave][k] = (_Float16)__expf(Af[wave][k] - mx);
        }
        __syncthreads();
        // ---- phase 2: WMMA  S = P @ EA  (via EAT-contiguous B fragments)
#pragma unroll
        for (int nt = wave * 2; nt < wave * 2 + 2; ++nt) {
            const int ncol = nt * 16 + nidx;
            const _Float16* btrow = BT + (size_t)ncol * KS;
            v8f acc = {};
#pragma unroll 4
            for (int kb = 0; kb < KS; kb += 32) {
                __builtin_prefetch(btrow + kb + 256, 0, 1);
                const _Float16* prow = &Pl[nidx][0];
                v8h alo = *(const v8h*)(prow + kb + half * 16);
                v8h ahi = *(const v8h*)(prow + kb + half * 16 + 8);
                v16h afrag;
#pragma unroll
                for (int e = 0; e < 8; ++e) { afrag[e] = alo[e]; afrag[e + 8] = ahi[e]; }
                v16h bfrag = *(const v16h*)(btrow + kb + half * 16);
                acc = __builtin_amdgcn_wmma_f32_16x16x32_f16(
                    false, afrag, false, bfrag, (short)0, acc, false, false);
            }
            // ---- phase 3: alpha_new = log(S) + rowmax + emission
#pragma unroll
            for (int r = 0; r < 8; ++r) {
                int   mrow = r + half * 8;
                int   b    = b0 + mrow;
                float xo   = obvs[(size_t)b * TT + t];
                float dd   = xo - s_mu[ncol];
                float em   = -0.5f * dd * dd * s_inv[ncol] + s_ct[ncol];
                float av   = __logf(acc[r]) + rowm[mrow] + em;
                Af[mrow][ncol] = av;
                alpha_out[(size_t)b * TT * KS + (size_t)t * KS + ncol] = av;
            }
        }
    }
}

// ---------------------------------------------------------------------------
// Backward pass + gamma: beta_t = log(exp(beta_{t+1}+e_{t+1}-m) @ exp(A)^T) + m
// gamma[b,t] = alpha + beta - LSE_k(alpha + beta), written in place over alpha.
// ---------------------------------------------------------------------------
__global__ __launch_bounds__(512) void hmm_backward(
    const float* __restrict__ obvs,
    const float* __restrict__ means,
    const float* __restrict__ log_stds,
    const _Float16* __restrict__ BT,    // EA row-major: [i][j] = exp(A[i][j])
    float* __restrict__ gout)           // holds alpha on entry, gamma on exit
{
    __shared__ float    Bf[BG][KS];     // current beta (fp32)
    __shared__ _Float16 Ql[BG][KS];     // exp(beta + e_next - rowmax)
    __shared__ float    rowm[BG];
    __shared__ float    s_mu[KS], s_inv[KS], s_ct[KS];

    const int tid  = threadIdx.x;
    const int wave = tid >> 5;
    const int lane = tid & 31;
    const int b0   = blockIdx.x * BG;
    const int half = lane >> 4;
    const int nidx = lane & 15;

    {
        float mu = means[tid], ls = log_stds[tid];
        s_mu[tid]  = mu;
        s_inv[tid] = __expf(-2.0f * ls);
        s_ct[tid]  = -ls - HALF_LOG_2PI;
    }
    // beta_{T-1} = 0
#pragma unroll 4
    for (int b = 0; b < BG; ++b) Bf[b][tid] = 0.0f;
    __syncthreads();

    // gamma at t = T-1 (wave w -> batch row w)
    {
        const size_t base = (size_t)(b0 + wave) * TT * KS + (size_t)(TT - 1) * KS;
        float v[KS / 32];
        float mx = NEG_INF;
#pragma unroll
        for (int i = 0; i < KS / 32; ++i) {
            float a = gout[base + lane + 32 * i] + Bf[wave][lane + 32 * i];
            v[i] = a; mx = fmaxf(mx, a);
        }
        mx = wave_max32(mx);
        float s = 0.0f;
#pragma unroll
        for (int i = 0; i < KS / 32; ++i) s += __expf(v[i] - mx);
        s = wave_sum32(s);
        float lse = mx + __logf(s);
#pragma unroll
        for (int i = 0; i < KS / 32; ++i) gout[base + lane + 32 * i] = v[i] - lse;
    }

    for (int t = TT - 2; t >= 0; --t) {
        __syncthreads();
        // ---- phase 1: Q = exp(beta_{t+1} + e_{t+1} - rowmax)
        {
            const float xo = obvs[(size_t)(b0 + wave) * TT + (t + 1)];
            float mx = NEG_INF;
            for (int k = lane; k < KS; k += 32) {
                float dd = xo - s_mu[k];
                float vv = Bf[wave][k] - 0.5f * dd * dd * s_inv[k] + s_ct[k];
                mx = fmaxf(mx, vv);
            }
            mx = wave_max32(mx);
            if (lane == 0) rowm[wave] = mx;
            for (int k = lane; k < KS; k += 32) {
                float dd = xo - s_mu[k];
                float vv = Bf[wave][k] - 0.5f * dd * dd * s_inv[k] + s_ct[k];
                Ql[wave][k] = (_Float16)__expf(vv - mx);
            }
        }
        __syncthreads();
        // ---- phase 2: WMMA  beta_t[b,i] = log(sum_j Q[b,j]*exp(A[i,j])) + m_b
#pragma unroll
        for (int nt = wave * 2; nt < wave * 2 + 2; ++nt) {
            const int ncol = nt * 16 + nidx;              // output state i
            const _Float16* btrow = BT + (size_t)ncol * KS; // EA[i][j], contiguous j
            v8f acc = {};
#pragma unroll 4
            for (int kb = 0; kb < KS; kb += 32) {
                __builtin_prefetch(btrow + kb + 256, 0, 1);
                const _Float16* qrow = &Ql[nidx][0];
                v8h alo = *(const v8h*)(qrow + kb + half * 16);
                v8h ahi = *(const v8h*)(qrow + kb + half * 16 + 8);
                v16h afrag;
#pragma unroll
                for (int e = 0; e < 8; ++e) { afrag[e] = alo[e]; afrag[e + 8] = ahi[e]; }
                v16h bfrag = *(const v16h*)(btrow + kb + half * 16);
                acc = __builtin_amdgcn_wmma_f32_16x16x32_f16(
                    false, afrag, false, bfrag, (short)0, acc, false, false);
            }
#pragma unroll
            for (int r = 0; r < 8; ++r) {
                int mrow = r + half * 8;
                Bf[mrow][ncol] = __logf(acc[r]) + rowm[mrow];
            }
        }
        __syncthreads();
        // ---- phase 3: gamma at t = alpha + beta - LSE, in place
        {
            const size_t base = (size_t)(b0 + wave) * TT * KS + (size_t)t * KS;
            float v[KS / 32];
            float mx = NEG_INF;
#pragma unroll
            for (int i = 0; i < KS / 32; ++i) {
                float a = gout[base + lane + 32 * i] + Bf[wave][lane + 32 * i];
                v[i] = a; mx = fmaxf(mx, a);
            }
            mx = wave_max32(mx);
            float s = 0.0f;
#pragma unroll
            for (int i = 0; i < KS / 32; ++i) s += __expf(v[i] - mx);
            s = wave_sum32(s);
            float lse = mx + __logf(s);
#pragma unroll
            for (int i = 0; i < KS / 32; ++i) gout[base + lane + 32 * i] = v[i] - lse;
        }
    }
}

// ---------------------------------------------------------------------------
extern "C" void kernel_launch(void* const* d_in, const int* in_sizes, int n_in,
                              void* d_out, int out_size, void* d_ws, size_t ws_size,
                              hipStream_t stream) {
    (void)in_sizes; (void)n_in; (void)out_size; (void)ws_size;
    const float* obvs      = (const float*)d_in[0];   // [B,T]
    const float* log_init  = (const float*)d_in[1];   // [K]
    const float* A         = (const float*)d_in[2];   // [K,K]
    const float* means     = (const float*)d_in[3];   // [K]
    const float* log_stds  = (const float*)d_in[4];   // [K]
    float* out = (float*)d_out;                        // [B,T,K]

    _Float16* EAT = (_Float16*)d_ws;                   // K*K f16
    _Float16* EA  = EAT + (size_t)KS * KS;             // K*K f16

    hmm_prep_expA<<<(KS * KS + 255) / 256, 256, 0, stream>>>(A, EAT, EA);
    hmm_forward<<<BB / BG, 512, 0, stream>>>(obvs, log_init, means, log_stds, EAT, out);
    hmm_backward<<<BB / BG, 512, 0, stream>>>(obvs, means, log_stds, EA, out);
}
